// LeVisionTransformer_20968030339770
// MI455X (gfx1250) — matrix operand order repaired and checked by reading, not compiled, not run on hardware
//
#include <hip/hip_runtime.h>

// ---------------------------------------------------------------------------
// LeViT-style transformer forward for MI455X (gfx1250, wave32, WMMA).
// GEMMs: v_wmma_f32_16x16x32_bf16, wave tile 16x64 (4 accumulators, A reuse),
// block tile 64x64 with the shared W tile staged into LDS via double-buffered
// global_load_async_to_lds_b128 (ASYNCcnt) — the CDNA5 async copy path.
// B fragments are hoisted so each k-step issues one ds_load clause + 4 WMMAs.
// ---------------------------------------------------------------------------

typedef unsigned short u16;
typedef __bf16 bf16_t;
typedef bf16_t v16bf __attribute__((ext_vector_type(16)));
typedef float  v8f   __attribute__((ext_vector_type(8)));
typedef u16    v8u   __attribute__((ext_vector_type(8)));

#define GKC 128   // K-chunk staged to LDS per double-buffer slot

__device__ __forceinline__ u16 f2bf(float f) {
  unsigned u = __builtin_bit_cast(unsigned, f);
  u += 0x7FFFu + ((u >> 16) & 1u);          // round-to-nearest-even
  return (u16)(u >> 16);
}

__device__ __forceinline__ float hswish_f(float x) {
  float t = x + 3.f;
  t = t < 0.f ? 0.f : (t > 6.f ? 6.f : t);
  return x * t * (1.f / 6.f);
}

__device__ __forceinline__ v8f zero8() {
  v8f z = {0.f, 0.f, 0.f, 0.f, 0.f, 0.f, 0.f, 0.f};
  return z;
}

// Build a v16bf fragment from two 16-byte half-vectors.
__device__ __forceinline__ v16bf make_frag(v8u lo, v8u hi) {
  v16bf r;
#pragma unroll
  for (int e = 0; e < 8; ++e) {
    r[e]     = __builtin_bit_cast(bf16_t, (u16)lo[e]);
    r[e + 8] = __builtin_bit_cast(bf16_t, (u16)hi[e]);
  }
  return r;
}
// A fragment (16x32): lane row base p (+k); elems 0..7 = K[kbA..kbA+7],
// elems 8..15 = K[16+kbA..16+kbA+7]; kbA = (lane<16)?0:8.
__device__ __forceinline__ v16bf frag_A(const u16* p, int kbA) {
  return make_frag(*(const v8u*)(p + kbA), *(const v8u*)(p + kbA + 16));
}
// B fragment (32x16) from row-major W[N,K] row: elems 0..7 = K[kbB..kbB+7],
// elems 8..15 = K[16+kbB..16+kbB+7]; kbB = (lane<16)?0:8.
__device__ __forceinline__ v16bf frag_B(const u16* p, int kbB) {
  return make_frag(*(const v8u*)(p + kbB), *(const v8u*)(p + kbB + 16));
}
__device__ __forceinline__ v8f wmma_acc(v16bf a, v16bf b, v8f c) {
  return __builtin_amdgcn_wmma_f32_16x16x32_bf16(false, a, false, b,
                                                 (short)0, c, false, false);
}

// Async-stage one 64-row x GKC-halves chunk of W[N,K] into LDS.
// 1024 16-byte transfers, 8 per thread (128 threads).
__device__ __forceinline__ void stage_async(u16* lds_base, const u16* __restrict__ W,
                                            int ldb, int N, int n0, int k0, int tid) {
#pragma unroll
  for (int q = 0; q < 8; ++q) {
    int id = q * 128 + tid;
    int row = id >> 4;          // 0..63
    int seg = id & 15;          // 16B segment within the 256B row chunk
    int rg = n0 + row; if (rg >= N) rg = N - 1;
    const u16* src = W + (size_t)rg * ldb + k0 + seg * 8;
    u16* dst = lds_base + row * GKC + seg * 8;
#if defined(__gfx1250__)
    unsigned doff = (unsigned)(unsigned long long)dst;   // LDS byte offset
    unsigned long long ga = (unsigned long long)src;
    asm volatile("global_load_async_to_lds_b128 %0, %1, off"
                 :: "v"(doff), "v"(ga) : "memory");
#else
    *(v8u*)dst = *(const v8u*)src;
#endif
  }
}

// ---------------------------------------------------------------------------
// Generic GEMM: C[M,N] = act( A[M,K](bf16) * W[N,K](bf16)^T * alpha
//                             + bias[n] + resid[m,n] )
// 128 threads = 4 waves. Block tile 64(M) x 64(N); wave tile 16x64.
// W tile double-buffered in LDS via async loads; counted s_wait_asynccnt 8
// lets the next chunk's DMA overlap this chunk's WMMAs.
// Requires K % 128 == 0 (all call sites satisfy this).
// ---------------------------------------------------------------------------
__global__ void gemm_kernel(const u16* __restrict__ A, int lda,
                            const u16* __restrict__ W, int ldb,
                            const float* __restrict__ bias,
                            const float* __restrict__ resid,
                            float* __restrict__ Cf, u16* __restrict__ Cbf,
                            int M, int N, int K, float alpha, int act) {
  __shared__ u16 Ws[2][64 * GKC];   // 32 KB
  int tid = threadIdx.x;
  int lane = tid & 31, wave = tid >> 5;
  int m0 = (blockIdx.y * 4 + wave) * 16;
  int n0 = blockIdx.x * 64;
  int mA = m0 + (lane & 15); if (mA >= M) mA = M - 1;
  int kbA = (lane < 16) ? 0 : 8;
  int kbB = (lane < 16) ? 0 : 8;
  const u16* Arow = A + (size_t)mA * lda;

  v8f acc[4];
#pragma unroll
  for (int t = 0; t < 4; ++t) acc[t] = zero8();

  stage_async(&Ws[0][0], W, ldb, N, n0, 0, tid);
  for (int kc = 0; kc < K; kc += GKC) {
    int buf = (kc / GKC) & 1;
    bool more = (kc + GKC) < K;
    if (more) stage_async(&Ws[buf ^ 1][0], W, ldb, N, n0, kc + GKC, tid);
#if defined(__gfx1250__)
    if (more) asm volatile("s_wait_asynccnt 8" ::: "memory");  // prev chunk done
    else      asm volatile("s_wait_asynccnt 0" ::: "memory");
#endif
    __syncthreads();
    const u16* Ap = Arow + kc;
    const u16* Bp = &Ws[buf][(lane & 15) * GKC];
#pragma unroll
    for (int kk = 0; kk < GKC; kk += 32) {
      v16bf af = frag_A(Ap + kk, kbA);
      v16bf b0 = frag_B(Bp + kk, kbB);
      v16bf b1 = frag_B(Bp + 16 * GKC + kk, kbB);
      v16bf b2 = frag_B(Bp + 32 * GKC + kk, kbB);
      v16bf b3 = frag_B(Bp + 48 * GKC + kk, kbB);
      acc[0] = wmma_acc(af, b0, acc[0]);
      acc[1] = wmma_acc(af, b1, acc[1]);
      acc[2] = wmma_acc(af, b2, acc[2]);
      acc[3] = wmma_acc(af, b3, acc[3]);
    }
    __syncthreads();
  }

  int mb = m0 + ((lane < 16) ? 0 : 8);
#pragma unroll
  for (int t = 0; t < 4; ++t) {
    int n = n0 + 16 * t + (lane & 15);
    if (n >= N) continue;
    float bn = bias ? bias[n] : 0.f;
#pragma unroll
    for (int r = 0; r < 8; ++r) {
      int m = mb + r;
      if (m < M) {
        float v = acc[t][r] * alpha + bn;
        if (resid) v += resid[(size_t)m * N + n];
        if (act == 1) v = hswish_f(v);
        if (Cf)  Cf [(size_t)m * N + n] = v;
        if (Cbf) Cbf[(size_t)m * N + n] = f2bf(v);
      }
    }
  }
}

// dots[b,h,i,j] = scale * sum_d q[b,h,i,d]*k[b,h,j,d] + biasTab[h,i,j]
// Wave tile 16(i) x 64(j), K=64 (two WMMA k-steps), 4 accumulators.
__global__ void attn_qk_kernel(const u16* __restrict__ Qbf, const u16* __restrict__ Kbf,
                               const float* __restrict__ biasTab, float* __restrict__ dots,
                               int H, int Tq, int T, int Tpad, int ik, float scale) {
  int lane = threadIdx.x & 31, wave = threadIdx.x >> 5;
  int bh = blockIdx.z;
  int b = bh / H, h = bh % H;
  int m0 = (blockIdx.y * 4 + wave) * 16;
  if (m0 >= Tq) return;                     // uniform per wave
  int n0 = blockIdx.x * 64;
  int mA = m0 + (lane & 15); if (mA >= Tq) mA = Tq - 1;
  int kbA = (lane < 16) ? 0 : 8;
  int kbB = (lane < 16) ? 0 : 8;
  const u16* Arow = Qbf + ((size_t)(b * Tq + mA)) * ik + h * 64;
  const u16* Brow[4];
#pragma unroll
  for (int t = 0; t < 4; ++t) {
    int nB = n0 + 16 * t + (lane & 15); if (nB >= T) nB = T - 1;
    Brow[t] = Kbf + ((size_t)(b * T + nB)) * ik + h * 64;
  }
  v8f acc[4];
#pragma unroll
  for (int t = 0; t < 4; ++t) acc[t] = zero8();
#pragma unroll
  for (int k = 0; k < 64; k += 32) {
    v16bf af = frag_A(Arow + k, kbA);
    v16bf b0 = frag_B(Brow[0] + k, kbB);
    v16bf b1 = frag_B(Brow[1] + k, kbB);
    v16bf b2 = frag_B(Brow[2] + k, kbB);
    v16bf b3 = frag_B(Brow[3] + k, kbB);
    acc[0] = wmma_acc(af, b0, acc[0]);
    acc[1] = wmma_acc(af, b1, acc[1]);
    acc[2] = wmma_acc(af, b2, acc[2]);
    acc[3] = wmma_acc(af, b3, acc[3]);
  }
  int mb = m0 + ((lane < 16) ? 0 : 8);
  float* drow = dots + (size_t)bh * Tq * Tpad;
  const float* btab = biasTab + (size_t)h * Tq * Tpad;
#pragma unroll
  for (int t = 0; t < 4; ++t) {
    int n = n0 + 16 * t + (lane & 15);
    if (n >= T) continue;
#pragma unroll
    for (int r = 0; r < 8; ++r) {
      int m = mb + r;
      if (m < Tq)
        drow[(size_t)m * Tpad + n] = acc[t][r] * scale + btab[(size_t)m * Tpad + n];
    }
  }
}

// out[b, i, h*64+d] = hswish( sum_j attn[b,h,i,j] * v[b,h,j,d] )   (bf16 out)
// Wave tile 16(i) x 64(d) — the whole head dim in one wave; K = Tpad.
__global__ void attn_av_kernel(const u16* __restrict__ attnbf, const u16* __restrict__ Vt,
                               u16* __restrict__ outbf, int H, int Tq, int Tpad, int ik) {
  int lane = threadIdx.x & 31, wave = threadIdx.x >> 5;
  int bh = blockIdx.z;
  int b = bh / H, h = bh % H;
  int m0 = (blockIdx.y * 4 + wave) * 16;
  if (m0 >= Tq) return;
  int mA = m0 + (lane & 15); if (mA >= Tq) mA = Tq - 1;
  int kbA = (lane < 16) ? 0 : 8;
  int kbB = (lane < 16) ? 0 : 8;
  const u16* Arow = attnbf + ((size_t)bh * Tq + mA) * Tpad;
  const u16* Brow[4];
#pragma unroll
  for (int t = 0; t < 4; ++t)
    Brow[t] = Vt + ((size_t)bh * 64 + 16 * t + (lane & 15)) * Tpad;
  v8f acc[4];
#pragma unroll
  for (int t = 0; t < 4; ++t) acc[t] = zero8();
  for (int k = 0; k < Tpad; k += 32) {
    v16bf af = frag_A(Arow + k, kbA);
    v16bf b0 = frag_B(Brow[0] + k, kbB);
    v16bf b1 = frag_B(Brow[1] + k, kbB);
    v16bf b2 = frag_B(Brow[2] + k, kbB);
    v16bf b3 = frag_B(Brow[3] + k, kbB);
    acc[0] = wmma_acc(af, b0, acc[0]);
    acc[1] = wmma_acc(af, b1, acc[1]);
    acc[2] = wmma_acc(af, b2, acc[2]);
    acc[3] = wmma_acc(af, b3, acc[3]);
  }
  int mb = m0 + ((lane < 16) ? 0 : 8);
#pragma unroll
  for (int t = 0; t < 4; ++t) {
    int n = 16 * t + (lane & 15);
#pragma unroll
    for (int r = 0; r < 8; ++r) {
      int m = mb + r;
      if (m < Tq)
        outbf[((size_t)(b * Tq + m)) * ik + h * 64 + n] = f2bf(hswish_f(acc[t][r]));
    }
  }
}

// ---------------------------------------------------------------------------
// Elementwise / helper kernels
// ---------------------------------------------------------------------------
__global__ void conv3x3s2(const float* __restrict__ in, const float* __restrict__ w,
                          const float* __restrict__ bias, float* __restrict__ out,
                          int B, int Ci, int Hi, int Wi, int Co, int Ho, int Wo) {
  size_t idx = (size_t)blockIdx.x * blockDim.x + threadIdx.x;
  size_t total = (size_t)B * Co * Ho * Wo;
  if (idx >= total) return;
  int wo = idx % Wo; size_t t = idx / Wo;
  int ho = t % Ho; t /= Ho;
  int co = t % Co; int b = t / Co;
  float acc = bias[co];
  int hi0 = ho * 2 - 1, wi0 = wo * 2 - 1;
  for (int ci = 0; ci < Ci; ++ci) {
    const float* ip = in + ((size_t)(b * Ci + ci) * Hi) * Wi;
    const float* wp = w + ((size_t)(co * Ci + ci)) * 9;
#pragma unroll
    for (int ky = 0; ky < 3; ++ky) {
      int hi = hi0 + ky;
      if (hi < 0 || hi >= Hi) continue;
#pragma unroll
      for (int kx = 0; kx < 3; ++kx) {
        int wi = wi0 + kx;
        if (wi < 0 || wi >= Wi) continue;
        acc += ip[(size_t)hi * Wi + wi] * wp[ky * 3 + kx];
      }
    }
  }
  out[idx] = acc;
}

__global__ void stem_resid(const float* __restrict__ img, const float* __restrict__ wr,
                           const float* __restrict__ br, float* __restrict__ x,
                           int B, int Co, int Ho, int Wo, int Hi, int Wi) {
  size_t idx = (size_t)blockIdx.x * blockDim.x + threadIdx.x;
  size_t total = (size_t)B * Co * Ho * Wo;
  if (idx >= total) return;
  int wo = idx % Wo; size_t t = idx / Wo;
  int ho = t % Ho; t /= Ho;
  int co = t % Co; int b = t / Co;
  float acc = br[co];
#pragma unroll
  for (int ci = 0; ci < 3; ++ci)
    acc += img[((size_t)(b * 3 + ci) * Hi + 16 * ho) * Wi + 16 * wo] * wr[co * 3 + ci];
  x[idx] += acc;
}

__global__ void tokenize_kernel(const float* __restrict__ x, float* __restrict__ tok,
                                int B, int C, int T) {
  size_t idx = (size_t)blockIdx.x * blockDim.x + threadIdx.x;
  size_t total = (size_t)B * T * C;
  if (idx >= total) return;
  int c = idx % C; size_t t2 = idx / C;
  int t = t2 % T; int b = t2 / T;
  tok[idx] = x[((size_t)(b * C + c)) * T + t];
}

__global__ void cvt_bf16(const float* __restrict__ src, u16* __restrict__ dst, size_t n) {
  size_t idx = (size_t)blockIdx.x * blockDim.x + threadIdx.x;
  if (idx < n) dst[idx] = f2bf(src[idx]);
}

__global__ void gather_q(const u16* __restrict__ xbf, u16* __restrict__ xq,
                         int B, int fh, int fw, int C) {
  int wq = (fw + 1) / 2, hq = (fh + 1) / 2;
  int Tq = wq * hq, T = fh * fw;
  size_t idx = (size_t)blockIdx.x * blockDim.x + threadIdx.x;
  size_t total = (size_t)B * Tq * C;
  if (idx >= total) return;
  int c = idx % C; size_t t2 = idx / C;
  int iq = t2 % Tq; int b = t2 / Tq;
  int qy = 2 * (iq / wq), qx = 2 * (iq % wq);
  xq[idx] = xbf[((size_t)b * T + qy * fw + qx) * C + c];
}

__global__ void fold_kernel(const float* __restrict__ w, const float* __restrict__ g,
                            const float* __restrict__ bb, const float* __restrict__ m,
                            const float* __restrict__ v, const float* __restrict__ cb,
                            u16* __restrict__ wb, float* __restrict__ bias, int N, int K) {
  size_t idx = (size_t)blockIdx.x * blockDim.x + threadIdx.x;
  if (idx >= (size_t)N * K) return;
  int n = idx / K;
  float s = g ? g[n] * rsqrtf(v[n] + 1e-5f) : 1.f;
  wb[idx] = f2bf(w[idx] * s);
  if ((idx % K) == 0) {
    float c0 = cb ? cb[n] : 0.f;
    bias[n] = g ? (c0 - m[n]) * s + bb[n] : c0;
  }
}

__global__ void bias_tab_kernel(const float* __restrict__ pos, float* __restrict__ tab,
                                int H, int fh, int fw, int ds, int Tq, int T, int Tpad,
                                float invscale) {
  size_t idx = (size_t)blockIdx.x * blockDim.x + threadIdx.x;
  size_t total = (size_t)H * Tq * Tpad;
  if (idx >= total) return;
  int j = idx % Tpad; size_t t2 = idx / Tpad;
  int i = t2 % Tq; int h = t2 / Tq;
  float val = 0.f;
  if (j < T) {
    int wq = ds ? (fw + 1) / 2 : fw;
    int step = ds ? 2 : 1;
    int qy = (i / wq) * step, qx = (i % wq) * step;
    int ky = j / fw, kx = j % fw;
    int ry = qy > ky ? qy - ky : ky - qy;
    int rx = qx > kx ? qx - kx : kx - qx;
    val = pos[(size_t)(ry * fw + rx) * H + h] * invscale;
  }
  tab[idx] = val;
}

__global__ void transposeV(const float* __restrict__ V, u16* __restrict__ Vt,
                           int B, int H, int T, int Tpad, int ik) {
  size_t idx = (size_t)blockIdx.x * blockDim.x + threadIdx.x;
  size_t total = (size_t)B * H * 64 * Tpad;
  if (idx >= total) return;
  int j = idx % Tpad; size_t t2 = idx / Tpad;
  int d = t2 % 64; int bh = t2 / 64;
  int b = bh / H, h = bh % H;
  float v = (j < T) ? V[((size_t)(b * T + j)) * ik + h * 64 + d] : 0.f;
  Vt[idx] = f2bf(v);
}

__global__ void softmax_kernel(const float* __restrict__ dots, u16* __restrict__ attnbf,
                               int T, int Tpad) {
  size_t row = blockIdx.x;
  const float* src = dots + row * (size_t)Tpad;
  u16* dst = attnbf + row * (size_t)Tpad;
  int lane = threadIdx.x;
  float mx = -1e30f;
  for (int j = lane; j < T; j += 32) mx = fmaxf(mx, src[j]);
  for (int o = 16; o; o >>= 1) mx = fmaxf(mx, __shfl_xor(mx, o));
  float sum = 0.f;
  for (int j = lane; j < T; j += 32) sum += __expf(src[j] - mx);
  for (int o = 16; o; o >>= 1) sum += __shfl_xor(sum, o);
  float inv = 1.f / sum;
  for (int j = lane; j < Tpad; j += 32)
    dst[j] = (j < T) ? f2bf(__expf(src[j] - mx) * inv) : (u16)0;
}

__global__ void pool_kernel(const float* __restrict__ x, u16* __restrict__ pooled,
                            int B, int T, int C) {
  int idx = blockIdx.x * blockDim.x + threadIdx.x;
  if (idx >= B * C) return;
  int c = idx % C, b = idx / C;
  float s = 0.f;
  for (int t = 0; t < T; ++t) s += x[((size_t)(b * T + t)) * C + c];
  pooled[idx] = f2bf(s / (float)T);
}

// ---------------------------------------------------------------------------
// Host orchestration
// ---------------------------------------------------------------------------
static inline unsigned cdiv(size_t n, unsigned b) { return (unsigned)((n + b - 1) / b); }

extern "C" void kernel_launch(void* const* d_in, const int* in_sizes, int n_in,
                              void* d_out, int out_size, void* d_ws, size_t ws_size,
                              hipStream_t stream) {
  (void)in_sizes; (void)n_in; (void)out_size;
  const size_t MB = 1024ull * 1024ull;
  if (ws_size < 289 * MB) return;  // workspace layout below requires ~288 MB

  const float* img = (const float*)d_in[0];
  auto F = [&](int i) { return (const float*)d_in[i]; };

  char* ws = (char*)d_ws;
  float* x1    = (float*)(ws + 0);          // 103 MB (stem) -> reused as dots
  float* x2    = (float*)(ws + 104 * MB);   // 52 MB (stem)  -> reused as attn bf16
  float* x3    = (float*)(ws + 156 * MB);   // 26 MB (stem)  -> reused as V f32
  float* x4    = (float*)(ws + 182 * MB);   // 13 MB stem output (NCHW)
  float* dots  = x1;
  u16*   attnb = (u16*)x2;
  float* Vf    = x3;
  float* xtokA = (float*)(ws + 195 * MB);   // 13 MB current x (token-major f32)
  float* xtokB = (float*)(ws + 208 * MB);   // 13 MB attention output x
  u16*   xbf   = (u16*)(ws + 221 * MB);     // 7 MB  bf16 activations
  u16*   Qbf   = (u16*)(ws + 228 * MB);     // 13 MB
  u16*   Kbf   = (u16*)(ws + 241 * MB);     // 13 MB
  u16*   Vt    = (u16*)(ws + 254 * MB);     // 15 MB V transposed [b,h,d,Tpad] bf16
  u16*   aux   = (u16*)(ws + 269 * MB);     // 13 MB attn-out bf16 / FF hidden bf16
  float* btab  = (float*)(ws + 282 * MB);   // 2 MB positional-bias table
  u16*   xqbf  = (u16*)(ws + 284 * MB);     // 2 MB gathered q-input (ds block)
  u16*   wfold = (u16*)(ws + 286 * MB);     // 1 MB folded bf16 weights
  float* bfold = (float*)(ws + 287 * MB);   // folded bias
  u16*   pooled = (u16*)(ws + 287 * MB + 64 * 1024);

  auto launch_gemm = [&](const u16* A, int lda, const u16* W, int ldb,
                         const float* bias, const float* resid,
                         float* Cf, u16* Cbf, int M, int N, int K,
                         float alpha, int act) {
    dim3 g((N + 63) / 64, (M + 63) / 64, 1);
    gemm_kernel<<<g, 128, 0, stream>>>(A, lda, W, ldb, bias, resid, Cf, Cbf,
                                       M, N, K, alpha, act);
  };

  // ---- Stem -------------------------------------------------------------
  {
    size_t n;
    n = 64ull * 32 * 112 * 112;
    conv3x3s2<<<cdiv(n, 256), 256, 0, stream>>>(img, F(1), F(2), x1, 64, 3, 224, 224, 32, 112, 112);
    n = 64ull * 64 * 56 * 56;
    conv3x3s2<<<cdiv(n, 256), 256, 0, stream>>>(x1, F(3), F(4), x2, 64, 32, 112, 112, 64, 56, 56);
    n = 64ull * 128 * 28 * 28;
    conv3x3s2<<<cdiv(n, 256), 256, 0, stream>>>(x2, F(5), F(6), x3, 64, 64, 56, 56, 128, 28, 28);
    n = 64ull * 256 * 14 * 14;
    conv3x3s2<<<cdiv(n, 256), 256, 0, stream>>>(x3, F(7), F(8), x4, 64, 128, 28, 28, 256, 14, 14);
    stem_resid<<<cdiv(n, 256), 256, 0, stream>>>(img, F(9), F(10), x4, 64, 256, 14, 14, 224, 224);
    tokenize_kernel<<<cdiv(n, 256), 256, 0, stream>>>(x4, xtokA, 64, 256, 196);
  }

  // ---- Transformer blocks ----------------------------------------------
  const int dims[3]  = {256, 256, 384};
  const int douts[3] = {256, 384, 384};
  const int heads[3] = {8, 8, 12};
  const int fhs[3]   = {14, 14, 7};
  const int dss[3]   = {0, 1, 0};
  const float scale = 0.125f;   // 64^-0.5

  float* cur = xtokA;   // block input  (token-major f32)
  float* tmp = xtokB;   // attention-stage output
  int Tcur = 196;

  for (int i = 0; i < 3; ++i) {
    const int base = 11 + 29 * i;   // wq index in flattened params dict
    const int C = dims[i], DO = douts[i], H = heads[i];
    const int fh = fhs[i], fw = fhs[i], ds = dss[i];
    const int T = fh * fw;
    const int Tq = ds ? ((fh + 1) / 2) * ((fw + 1) / 2) : T;
    const int Tpad = ((T + 31) / 32) * 32;      // K-padding for attn*V
    const int ik = H * 64, hid = DO * 2;
    const int Mi = 64 * T, Mq = 64 * Tq;

    // bf16 copy of x
    cvt_bf16<<<cdiv((size_t)Mi * C, 256), 256, 0, stream>>>(cur, xbf, (size_t)Mi * C);
    // positional bias table [h, i, j] (already divided by scale)
    bias_tab_kernel<<<cdiv((size_t)H * Tq * Tpad, 256), 256, 0, stream>>>(
        F(base + 21), btab, H, fh, fw, ds, Tq, T, Tpad, 8.0f);

    // Q (optionally on strided q-token subset)
    const u16* Aq = xbf;
    if (ds) {
      gather_q<<<cdiv((size_t)Mq * C, 256), 256, 0, stream>>>(xbf, xqbf, 64, fh, fw, C);
      Aq = xqbf;
    }
    fold_kernel<<<cdiv((size_t)ik * C, 256), 256, 0, stream>>>(
        F(base + 0), F(base + 1), F(base + 2), F(base + 3), F(base + 4), nullptr,
        wfold, bfold, ik, C);
    launch_gemm(Aq, C, wfold, C, bfold, nullptr, nullptr, Qbf, Mq, ik, C, 1.f, 0);

    // K
    fold_kernel<<<cdiv((size_t)ik * C, 256), 256, 0, stream>>>(
        F(base + 5), F(base + 6), F(base + 7), F(base + 8), F(base + 9), nullptr,
        wfold, bfold, ik, C);
    launch_gemm(xbf, C, wfold, C, bfold, nullptr, nullptr, Kbf, Mi, ik, C, 1.f, 0);

    // V (f32, then transpose to [b,h,d,Tpad] bf16)
    fold_kernel<<<cdiv((size_t)ik * C, 256), 256, 0, stream>>>(
        F(base + 10), F(base + 11), F(base + 12), F(base + 13), F(base + 14), nullptr,
        wfold, bfold, ik, C);
    launch_gemm(xbf, C, wfold, C, bfold, nullptr, Vf, nullptr, Mi, ik, C, 1.f, 0);
    transposeV<<<cdiv((size_t)64 * H * 64 * Tpad, 256), 256, 0, stream>>>(
        Vf, Vt, 64, H, T, Tpad, ik);

    // dots = scale*q.k^T + bias; softmax; out = attn.v (hswish fused)
    {
      dim3 gqk((T + 63) / 64, (Tq + 63) / 64, 64 * H);
      attn_qk_kernel<<<gqk, 128, 0, stream>>>(Qbf, Kbf, btab, dots, H, Tq, T, Tpad, ik, scale);
      softmax_kernel<<<dim3((unsigned)(64 * H * Tq)), 32, 0, stream>>>(dots, attnb, T, Tpad);
      dim3 gav(1, (Tq + 63) / 64, 64 * H);
      attn_av_kernel<<<gav, 128, 0, stream>>>(attnb, Vt, aux, H, Tq, Tpad, ik);
    }

    // output projection (conv bias bo folded with bno) + residual
    fold_kernel<<<cdiv((size_t)DO * ik, 256), 256, 0, stream>>>(
        F(base + 15), F(base + 17), F(base + 18), F(base + 19), F(base + 20),
        F(base + 16), wfold, bfold, DO, ik);
    const float* resid = (!ds && C == DO) ? cur : nullptr;
    launch_gemm(aux, ik, wfold, ik, bfold, resid, tmp, nullptr, Mq, DO, ik, 1.f, 0);

    // FFN: hswish(bn(w1 x)) then w2 + b2 + residual
    cvt_bf16<<<cdiv((size_t)Mq * DO, 256), 256, 0, stream>>>(tmp, xbf, (size_t)Mq * DO);
    fold_kernel<<<cdiv((size_t)hid * DO, 256), 256, 0, stream>>>(
        F(base + 22), F(base + 23), F(base + 24), F(base + 25), F(base + 26), nullptr,
        wfold, bfold, hid, DO);
    launch_gemm(xbf, DO, wfold, DO, bfold, nullptr, nullptr, aux, Mq, hid, DO, 1.f, 1);
    fold_kernel<<<cdiv((size_t)DO * hid, 256), 256, 0, stream>>>(
        F(base + 27), nullptr, nullptr, nullptr, nullptr, F(base + 28),
        wfold, bfold, DO, hid);
    launch_gemm(aux, hid, wfold, hid, bfold, tmp, cur, nullptr, Mq, DO, hid, 1.f, 0);

    Tcur = Tq;  // cur now holds [64*Tq, DO]
  }

  // ---- Pool + head -------------------------------------------------------
  pool_kernel<<<cdiv((size_t)64 * 384, 256), 256, 0, stream>>>(cur, pooled, 64, Tcur, 384);
  fold_kernel<<<cdiv((size_t)1000 * 384, 256), 256, 0, stream>>>(
      F(98), nullptr, nullptr, nullptr, nullptr, F(99), wfold, bfold, 1000, 384);
  launch_gemm(pooled, 384, wfold, 384, bfold, nullptr, (float*)d_out, nullptr,
              64, 1000, 384, 1.f, 0);
}
